// HeteroGNNModel_50732153700723
// MI455X (gfx1250) — matrix operand order repaired and checked by reading, not compiled, not run on hardware
//
#include <hip/hip_runtime.h>
#include <hip/hip_bf16.h>
#include <math.h>

#define NU   20000
#define NF   1000
#define EUF  100000
#define EUU  200000
#define HIDD 128

typedef _Float16 half_t;
typedef __attribute__((ext_vector_type(16))) _Float16 v16h;
typedef __attribute__((ext_vector_type(8)))  float    v8f;
typedef __attribute__((ext_vector_type(4)))  float    f32x4;

// ---------------------------------------------------------------- utilities

__device__ __forceinline__ void atomicMaxF(float* addr, float val) {
  // monotonic int/uint trick; buffer initialized to -inf
  if (val >= 0.f) atomicMax((int*)addr, __float_as_int(val));
  else            atomicMin((unsigned int*)addr, __float_as_uint(val));
}

__global__ void fill_kernel(float* __restrict__ p, long n, float v) {
  long i = (long)blockIdx.x * blockDim.x + threadIdx.x;
  if (i < n) p[i] = v;
}

// ---------------------------------------------------------------- WMMA GEMM
// C[M,N] = act( A[M,K] @ B[K,N] + bias + addvec ), f32 in/out, f16 WMMA math.
// Requires K % 32 == 0, N % 16 == 0. Rows are clamped (not predicated) so all
// loads are unconditional; stores are guarded.
//
// A-tile (16x32 f16, ISA 7.12.2): lane L -> row = L%16; element e maps to
//   k = k0 + hi*8 + e           (e = 0..7)
//   k = k0 + hi*8 + 16 + (e-8)  (e = 8..15)    where hi = L>>4
// i.e. two contiguous 8-float runs -> four b128 loads.
// B-tile (32x16 f16): lane L -> col = L%16; element e -> same k as A-tile.
// C/D (8 VGPR f32): vgpr i -> row = i + hi*8, col = L%16.

__device__ __forceinline__ v16h load_a16(const float* __restrict__ A, long stride,
                                         int row, int koff, int hi) {
  const float* p = A + (long)row * stride + koff + hi * 8;
  f32x4 f0 = *(const f32x4*)(p);
  f32x4 f1 = *(const f32x4*)(p + 4);
  f32x4 f2 = *(const f32x4*)(p + 16);
  f32x4 f3 = *(const f32x4*)(p + 20);
  v16h a;
#pragma unroll
  for (int j = 0; j < 4; ++j) {
    a[j]      = (half_t)f0[j];
    a[4 + j]  = (half_t)f1[j];
    a[8 + j]  = (half_t)f2[j];
    a[12 + j] = (half_t)f3[j];
  }
  return a;
}

__device__ __forceinline__ v16h load_b16(const float* __restrict__ B, int N,
                                         int k0, int col, int hi) {
  const float* p = B + (long)(k0 + hi * 8) * N + col;
  v16h b;
#pragma unroll
  for (int e = 0; e < 16; ++e) b[e] = (half_t)p[(long)(e + (e & 8)) * N];
  return b;
}

__device__ __forceinline__ v8f wmma_f16(v16h a, v16h b, v8f c) {
  return __builtin_amdgcn_wmma_f32_16x16x32_f16(false, a, false, b, (short)0, c,
                                                false, false);
}

__global__ __launch_bounds__(128) void gemm_wmma(
    const float* __restrict__ A, const float* __restrict__ B,
    const float* __restrict__ bias, const float* __restrict__ addvec,
    float* __restrict__ C, int M, int N, int K, int relu) {
  int lane = threadIdx.x & 31;
  int wave = threadIdx.x >> 5;
  int n0 = (blockIdx.x * 4 + wave) * 16;   // 4 waves/block, 16 cols each
  int m0 = blockIdx.y * 32;                // 2 row-tiles per wave
  if (n0 >= N) return;                     // uniform per wave
  int hi  = lane >> 4;
  int col = n0 + (lane & 15);
  int r0 = m0 + (lane & 15);       r0 = (r0 < M) ? r0 : M - 1;
  int r1 = m0 + 16 + (lane & 15);  r1 = (r1 < M) ? r1 : M - 1;
  v8f c0 = {}, c1 = {};
  for (int k0 = 0; k0 < K; k0 += 32) {
    v16h b  = load_b16(B, N, k0, col, hi);
    v16h a0 = load_a16(A, K, r0, k0, hi);
    v16h a1 = load_a16(A, K, r1, k0, hi);
    c0 = wmma_f16(a0, b, c0);
    c1 = wmma_f16(a1, b, c1);
  }
  float badd = 0.f;
  if (bias)   badd += bias[col];
  if (addvec) badd += addvec[col];
#pragma unroll
  for (int i = 0; i < 8; ++i) {
    int row = m0 + i + hi * 8;
    if (row < M) {
      float v = c0[i] + badd;
      if (relu) v = fmaxf(v, 0.f);
      C[(long)row * N + col] = v;
    }
    int rowb = row + 16;
    if (rowb < M) {
      float v = c1[i] + badd;
      if (relu) v = fmaxf(v, 0.f);
      C[(long)rowb * N + col] = v;
    }
  }
}

// GEMM over virtual concat A = [user_h(128) | g_broadcast(128) | dsl_embed(128)].
// K = 384; each 32-wide K-block lies entirely inside one 128-aligned segment,
// so the source pointer is selected once per k0 (wave-uniform branch).
__global__ __launch_bounds__(128) void gemm_cat_wmma(
    const float* __restrict__ uh, const float* __restrict__ gv,
    const float* __restrict__ de, const float* __restrict__ B,
    const float* __restrict__ bias, float* __restrict__ C, int M, int N) {
  const int K = 384;
  int lane = threadIdx.x & 31;
  int wave = threadIdx.x >> 5;
  int n0 = (blockIdx.x * 4 + wave) * 16;
  int m0 = blockIdx.y * 32;
  if (n0 >= N) return;
  int hi  = lane >> 4;
  int col = n0 + (lane & 15);
  int r0 = m0 + (lane & 15);       r0 = (r0 < M) ? r0 : M - 1;
  int r1 = m0 + 16 + (lane & 15);  r1 = (r1 < M) ? r1 : M - 1;
  v8f c0 = {}, c1 = {};
  for (int k0 = 0; k0 < K; k0 += 32) {
    const float* P; long stride; int koff;
    if (k0 < 128)      { P = uh; stride = 128; koff = k0; }
    else if (k0 < 256) { P = gv; stride = 0;   koff = k0 - 128; }
    else               { P = de; stride = 128; koff = k0 - 256; }
    v16h b  = load_b16(B, N, k0, col, hi);
    v16h a0 = load_a16(P, stride, r0, koff, hi);
    v16h a1 = load_a16(P, stride, r1, koff, hi);
    c0 = wmma_f16(a0, b, c0);
    c1 = wmma_f16(a1, b, c1);
  }
  float badd = bias[col];
#pragma unroll
  for (int i = 0; i < 8; ++i) {
    int row = m0 + i + hi * 8;
    if (row < M) C[(long)row * N + col] = fmaxf(c0[i] + badd, 0.f);
    int rowb = row + 16;
    if (rowb < M) C[(long)rowb * N + col] = fmaxf(c1[i] + badd, 0.f);
  }
}

// ---------------------------------------------------------------- GNN pieces

// a_s[n,h] = <h[n,h,:], att_src[h,:]>, a_d likewise
__global__ void node_att(const float* __restrict__ h, const float* __restrict__ as,
                         const float* __restrict__ ad, float* __restrict__ nas,
                         float* __restrict__ nad, int N, int H) {
  int i = blockIdx.x * blockDim.x + threadIdx.x;
  if (i >= N * H) return;
  int n = i / H, hh = i % H;
  const float* hp  = h + ((long)n * H + hh) * HIDD;
  const float* asp = as + hh * HIDD;
  const float* adp = ad + hh * HIDD;
  float s = 0.f, d = 0.f;
  for (int c = 0; c < HIDD; ++c) { float v = hp[c]; s += v * asp[c]; d += v * adp[c]; }
  nas[i] = s; nad[i] = d;
}

// generic GAT over src/dst edge list + implicit self loops (e >= E)
__global__ void gat_pass1(const int* __restrict__ src, const int* __restrict__ dst,
                          int E, int Etot, int H, const float* __restrict__ nas,
                          const float* __restrict__ nad, float* __restrict__ score,
                          float* __restrict__ nm) {
  int i = blockIdx.x * blockDim.x + threadIdx.x;
  if (i >= Etot * H) return;
  int e = i / H, hh = i % H;
  int s = (e < E) ? src[e] : (e - E);
  int d = (e < E) ? dst[e] : (e - E);
  float sc = nas[s * H + hh] + nad[d * H + hh];
  sc = (sc > 0.f) ? sc : 0.2f * sc;          // leaky_relu 0.2
  score[i] = sc;
  atomicMaxF(&nm[d * H + hh], sc);
}

__global__ void gat_pass2(const int* __restrict__ dst, int E, int Etot, int H,
                          float* __restrict__ score, const float* __restrict__ nm,
                          float* __restrict__ nden) {
  int i = blockIdx.x * blockDim.x + threadIdx.x;
  if (i >= Etot * H) return;
  int e = i / H, hh = i % H;
  int d = (e < E) ? dst[e] : (e - E);
  float ex = expf(score[i] - nm[d * H + hh]);
  score[i] = ex;
  atomicAdd(&nden[d * H + hh], ex);
}

__global__ void gat_pass3(const int* __restrict__ src, const int* __restrict__ dst,
                          int E, int Etot, int H, const float* __restrict__ h,
                          const float* __restrict__ score, const float* __restrict__ nden,
                          float* __restrict__ out) {
  int i = blockIdx.x * blockDim.x + threadIdx.x;
  if (i >= Etot * HIDD) return;
  int e = i >> 7, c = i & 127;
  int s = (e < E) ? src[e] : (e - E);
  int d = (e < E) ? dst[e] : (e - E);
  float acc = 0.f;
  for (int hh = 0; hh < H; ++hh) {
    float al = score[e * H + hh] / (nden[d * H + hh] + 1e-16f);
    acc += al * h[((long)s * H + hh) * HIDD + c];
  }
  atomicAdd(&out[(long)d * HIDD + c], acc / (float)H);
}

// relational (feature->user) GAT
__global__ void rel_pass1(const int* __restrict__ hu, const int* __restrict__ hf,
                          const float* __restrict__ hdst, const float* __restrict__ hsrc,
                          const float* __restrict__ ra, float* __restrict__ score,
                          float* __restrict__ nm) {
  int i = blockIdx.x * blockDim.x + threadIdx.x;
  if (i >= EUF * 4) return;
  int e = i >> 2, hh = i & 3;
  int u = hu[e], f = hf[e];
  const float* xi = hdst + ((long)u * 4 + hh) * HIDD;
  const float* xj = hsrc + ((long)f * 4 + hh) * HIDD;
  const float* r  = ra + hh * 256;
  float s = 0.f;
  for (int c = 0; c < HIDD; ++c) s += xi[c] * r[c] + xj[c] * r[128 + c];
  s = (s > 0.f) ? s : 0.2f * s;
  score[i] = s;
  atomicMaxF(&nm[u * 4 + hh], s);
}

__global__ void rel_pass2(const int* __restrict__ hu, float* __restrict__ score,
                          const float* __restrict__ nm, float* __restrict__ nden) {
  int i = blockIdx.x * blockDim.x + threadIdx.x;
  if (i >= EUF * 4) return;
  int e = i >> 2, hh = i & 3;
  int u = hu[e];
  float ex = expf(score[i] - nm[u * 4 + hh]);
  score[i] = ex;
  atomicAdd(&nden[u * 4 + hh], ex);
}

__global__ void rel_pass3(const int* __restrict__ hu, const int* __restrict__ hf,
                          const float* __restrict__ attr, const float* __restrict__ hdst,
                          const float* __restrict__ We, const float* __restrict__ Wz,
                          const float* __restrict__ score, const float* __restrict__ nden,
                          float* __restrict__ out) {
  int i = blockIdx.x * blockDim.x + threadIdx.x;
  if (i >= EUF * HIDD) return;
  int e = i >> 7, c = i & 127;
  int u = hu[e];
  float a0 = attr[e * 2 + 0], a1 = attr[e * 2 + 1];
  float wz = a1 * Wz[c];
  float acc = 0.f;
  for (int hh = 0; hh < 4; ++hh) {
    float ew = 1.f / (1.f + expf(-(a0 * We[hh] + a1 * We[4 + hh])));
    float al = score[e * 4 + hh] / (nden[u * 4 + hh] + 1e-16f) * ew;
    float msg = a0 * (hdst[((long)u * 4 + hh) * HIDD + c] + wz);
    acc += al * msg;
  }
  atomicAdd(&out[(long)u * HIDD + c], acc * 0.25f);
}

__global__ void finalize_node(const float* __restrict__ acc, const float* __restrict__ b1,
                              const float* __restrict__ b2, float* __restrict__ out,
                              int n, int relu) {
  int i = blockIdx.x * blockDim.x + threadIdx.x;
  if (i >= n) return;
  int c = i & 127;
  float v = acc[i] + b1[c] + (b2 ? b2[c] : 0.f);
  if (relu) v = fmaxf(v, 0.f);
  out[i] = v;
}

// ---------------------------------------------------------------- attention pool

__global__ void dotrow(const float* __restrict__ Q, const float* __restrict__ Km,
                       float* __restrict__ s) {
  int n = blockIdx.x * blockDim.x + threadIdx.x;
  if (n >= NU) return;
  float acc = 0.f;
  const float* q = Q + (long)n * HIDD;
  const float* k = Km + (long)n * HIDD;
  for (int c = 0; c < HIDD; ++c) acc += q[c] * k[c];
  s[n] = acc * 0.08838834764831845f;  // 1/sqrt(128)
}

__global__ void reduce_max_k(const float* __restrict__ s, int n, float* __restrict__ out) {
  __shared__ float red[256];
  float m = -__builtin_inff();
  for (int i = blockIdx.x * blockDim.x + threadIdx.x; i < n; i += gridDim.x * blockDim.x)
    m = fmaxf(m, s[i]);
  red[threadIdx.x] = m; __syncthreads();
  for (int st = 128; st > 0; st >>= 1) {
    if ((int)threadIdx.x < st) red[threadIdx.x] = fmaxf(red[threadIdx.x], red[threadIdx.x + st]);
    __syncthreads();
  }
  if (threadIdx.x == 0) atomicMaxF(out, red[0]);
}

__global__ void expsum_k(float* __restrict__ s, int n, const float* __restrict__ mx,
                         float* __restrict__ sum) {
  int i = blockIdx.x * blockDim.x + threadIdx.x;
  if (i >= n) return;
  float w = expf(s[i] - mx[0]);
  s[i] = w;
  atomicAdd(sum, w);
}

__global__ void gpool_k(const float* __restrict__ w, const float* __restrict__ V,
                        const float* __restrict__ sum, float* __restrict__ g) {
  __shared__ float red[256];
  int c = blockIdx.x;
  float s = 0.f;
  for (int n = threadIdx.x; n < NU; n += 256) s += w[n] * V[(long)n * HIDD + c];
  red[threadIdx.x] = s; __syncthreads();
  for (int st = 128; st > 0; st >>= 1) {
    if ((int)threadIdx.x < st) red[threadIdx.x] += red[threadIdx.x + st];
    __syncthreads();
  }
  if (threadIdx.x == 0) g[c] = red[0] / sum[0];
}

__global__ void matvec128(const float* __restrict__ g, const float* __restrict__ W,
                          const float* __restrict__ b, float* __restrict__ out) {
  int c = threadIdx.x;  // 128 threads
  float s = b[c];
  for (int k = 0; k < HIDD; ++k) s += g[k] * W[k * HIDD + c];
  out[c] = s;
}

// ---------------------------------------------------------------- DSL branch

__global__ void scatter_dsl(const int* __restrict__ hu, const int* __restrict__ hf,
                            const float* __restrict__ attr, float* __restrict__ dsl) {
  int e = blockIdx.x * blockDim.x + threadIdx.x;
  if (e >= EUF) return;
  dsl[(long)hu[e] * NF + hf[e]] = 1.f - attr[e * 2];
}

// Conv1d(1->16,k=3,p=1) + mean over length collapses to 3 taps of the row sum.
// dslm is [NU,32], upper 16 channels stay zero (K padded to 32 for WMMA GEMM).
__global__ void dsl_rowconv(const float* __restrict__ dsl, const float* __restrict__ ck,
                            const float* __restrict__ cb, float* __restrict__ dslm) {
  __shared__ float red[256];
  int u = blockIdx.x;
  const float* row = dsl + (long)u * NF;
  float s = 0.f;
  for (int f = threadIdx.x; f < NF; f += 256) s += row[f];
  red[threadIdx.x] = s; __syncthreads();
  for (int st = 128; st > 0; st >>= 1) {
    if ((int)threadIdx.x < st) red[threadIdx.x] += red[threadIdx.x + st];
    __syncthreads();
  }
  if (threadIdx.x < 16) {
    float S = red[0], d0 = row[0], dL = row[NF - 1];
    int oc = threadIdx.x;
    float v = ck[oc * 3 + 0] * (S - dL) + ck[oc * 3 + 1] * S + ck[oc * 3 + 2] * (S - d0);
    dslm[u * 32 + oc] = cb[oc] + v * (1.0f / (float)NF);
  }
}

// ---------------------------------------------------------------- host driver

static void launch_gemm(const float* A, const float* B, const float* bias,
                        const float* addvec, float* C, int M, int N, int K, int relu,
                        hipStream_t st) {
  dim3 grid((N / 16 + 3) / 4, (M + 31) / 32);
  gemm_wmma<<<grid, 128, 0, st>>>(A, B, bias, addvec, C, M, N, K, relu);
}

static void launch_fill(float* p, long n, float v, hipStream_t st) {
  fill_kernel<<<(int)((n + 255) / 256), 256, 0, st>>>(p, n, v);
}

extern "C" void kernel_launch(void* const* d_in, const int* in_sizes, int n_in,
                              void* d_out, int out_size, void* d_ws, size_t ws_size,
                              hipStream_t stream) {
  const float* user_x = (const float*)d_in[0];
  const float* feat_e = (const float*)d_in[1];
  const int*   has_u  = (const int*)d_in[2];
  const int*   has_f  = (const int*)d_in[3];
  const float* attr   = (const float*)d_in[4];
  const int*   s_src  = (const int*)d_in[5];
  const int*   s_dst  = (const int*)d_in[6];
  const float *Wu = (const float*)d_in[7],  *bu = (const float*)d_in[8];
  const float *Wf = (const float*)d_in[9],  *bf = (const float*)d_in[10];
  const float *rWsrc = (const float*)d_in[11], *rWdst = (const float*)d_in[12];
  const float *rWe = (const float*)d_in[13], *rWz = (const float*)d_in[14];
  const float *ra  = (const float*)d_in[15], *rb  = (const float*)d_in[16];
  const float *gW  = (const float*)d_in[17], *gas = (const float*)d_in[18];
  const float *gad = (const float*)d_in[19], *gb  = (const float*)d_in[20];
  const float *c1W = (const float*)d_in[21], *c1as = (const float*)d_in[22];
  const float *c1ad = (const float*)d_in[23], *c1b = (const float*)d_in[24];
  const float *c2W = (const float*)d_in[25], *c2as = (const float*)d_in[26];
  const float *c2ad = (const float*)d_in[27], *c2b = (const float*)d_in[28];
  const float *Wq = (const float*)d_in[29], *bq = (const float*)d_in[30];
  const float *Wk = (const float*)d_in[31], *bk = (const float*)d_in[32];
  const float *Wv = (const float*)d_in[33], *bv = (const float*)d_in[34];
  const float *Wm = (const float*)d_in[35], *bm = (const float*)d_in[36];
  const float *Wn = (const float*)d_in[37], *bn = (const float*)d_in[38];
  const float *ck = (const float*)d_in[39], *cb = (const float*)d_in[40];
  const float *Wd = (const float*)d_in[41], *bd = (const float*)d_in[42];
  const float *Wo1 = (const float*)d_in[43], *bo1 = (const float*)d_in[44];
  const float *Wo2 = (const float*)d_in[45], *bo2 = (const float*)d_in[46];
  float* out = (float*)d_out;

  // bump allocator over workspace (floats, 1KB-aligned chunks)
  float* W = (float*)d_ws;
  size_t off = 0;
  auto alloc = [&](size_t n) { float* p = W + off; off += (n + 255) & ~(size_t)255; return p; };
  float* big   = alloc((size_t)NU * NF);   // dsl dense; later h[NU,512]; later Q/K/V
  float* xu    = alloc((size_t)NU * HIDD);
  float* xf    = alloc((size_t)NF * HIDD);
  float* hsrc  = alloc((size_t)NF * 4 * HIDD);
  float* acc   = alloc((size_t)NU * HIDD);
  float* user  = alloc((size_t)NU * HIDD);
  float* h1    = alloc((size_t)NU * HIDD);
  float* h2    = alloc((size_t)NU * HIDD);
  float* userh = alloc((size_t)NU * HIDD);
  float* dslE  = alloc((size_t)NU * HIDD);
  float* dslm  = alloc((size_t)NU * 32);   // K padded to 32 (upper half zero)
  float* esc   = alloc((size_t)(EUU + NU) * 4);
  float* nm    = alloc((size_t)NU * 4);
  float* nden  = alloc((size_t)NU * 4);
  float* nas   = alloc((size_t)NU * 4);
  float* nad   = alloc((size_t)NU * 4);
  float* sw    = alloc(NU);
  float* gvec  = alloc(128);
  float* gn    = alloc(128);
  float* smax  = alloc(2);                 // [0]=max, [1]=sum
  float* out1  = alloc((size_t)NU * 64);

  const int Etot = EUU + NU;               // social edges + self loops
  const int TPB = 256;

  // ---- Phase 0: DSL branch (uses `big` as dense [NU,NF]) ----
  launch_fill(big, (long)NU * NF, 0.f, stream);
  launch_fill(dslm, (long)NU * 32, 0.f, stream);
  scatter_dsl<<<(EUF + TPB - 1) / TPB, TPB, 0, stream>>>(has_u, has_f, attr, big);
  dsl_rowconv<<<NU, 256, 0, stream>>>(big, ck, cb, dslm);
  launch_gemm(dslm, Wd, bd, nullptr, dslE, NU, HIDD, 32, 0, stream);

  // ---- Phase 1: projections ----
  launch_gemm(user_x, Wu, bu, nullptr, xu, NU, HIDD, 256, 0, stream);
  launch_gemm(feat_e, Wf, bf, nullptr, xf, NF, HIDD, 32, 0, stream);

  // ---- Phase 2: relational GAT (feature -> user), `big` = h_dst [NU, 4*128] ----
  launch_gemm(xu, rWdst, nullptr, nullptr, big, NU, 4 * HIDD, HIDD, 0, stream);
  launch_gemm(xf, rWsrc, nullptr, nullptr, hsrc, NF, 4 * HIDD, HIDD, 0, stream);
  launch_fill(nm, (long)NU * 4, -__builtin_inff(), stream);
  launch_fill(nden, (long)NU * 4, 0.f, stream);
  launch_fill(acc, (long)NU * HIDD, 0.f, stream);
  rel_pass1<<<(EUF * 4 + TPB - 1) / TPB, TPB, 0, stream>>>(has_u, has_f, big, hsrc, ra, esc, nm);
  rel_pass2<<<(EUF * 4 + TPB - 1) / TPB, TPB, 0, stream>>>(has_u, esc, nm, nden);
  rel_pass3<<<(EUF * HIDD + TPB - 1) / TPB, TPB, 0, stream>>>(has_u, has_f, attr, big, rWe, rWz,
                                                              esc, nden, acc);

  // ---- Phase 3: social GAT (H=4, self loops), `big` = h [NU, 4*128] ----
  launch_gemm(xu, gW, nullptr, nullptr, big, NU, 4 * HIDD, HIDD, 0, stream);
  node_att<<<(NU * 4 + TPB - 1) / TPB, TPB, 0, stream>>>(big, gas, gad, nas, nad, NU, 4);
  launch_fill(nm, (long)NU * 4, -__builtin_inff(), stream);
  launch_fill(nden, (long)NU * 4, 0.f, stream);
  gat_pass1<<<(Etot * 4 + TPB - 1) / TPB, TPB, 0, stream>>>(s_src, s_dst, EUU, Etot, 4, nas, nad, esc, nm);
  gat_pass2<<<(Etot * 4 + TPB - 1) / TPB, TPB, 0, stream>>>(s_dst, EUU, Etot, 4, esc, nm, nden);
  gat_pass3<<<(Etot * HIDD + TPB - 1) / TPB, TPB, 0, stream>>>(s_src, s_dst, EUU, Etot, 4, big, esc, nden, acc);
  finalize_node<<<(NU * HIDD + TPB - 1) / TPB, TPB, 0, stream>>>(acc, rb, gb, user, NU * HIDD, 1);

  // ---- Phase 4: CauGramer GAT layers (H=1) ----
  for (int layer = 0; layer < 2; ++layer) {
    const float* x   = (layer == 0) ? user : h1;
    const float* Wl  = (layer == 0) ? c1W : c2W;
    const float* asl = (layer == 0) ? c1as : c2as;
    const float* adl = (layer == 0) ? c1ad : c2ad;
    const float* bl  = (layer == 0) ? c1b : c2b;
    float* dst       = (layer == 0) ? h1 : h2;
    launch_gemm(x, Wl, nullptr, nullptr, big, NU, HIDD, HIDD, 0, stream);
    node_att<<<(NU + TPB - 1) / TPB, TPB, 0, stream>>>(big, asl, adl, nas, nad, NU, 1);
    launch_fill(nm, NU, -__builtin_inff(), stream);
    launch_fill(nden, NU, 0.f, stream);
    launch_fill(acc, (long)NU * HIDD, 0.f, stream);
    gat_pass1<<<(Etot + TPB - 1) / TPB, TPB, 0, stream>>>(s_src, s_dst, EUU, Etot, 1, nas, nad, esc, nm);
    gat_pass2<<<(Etot + TPB - 1) / TPB, TPB, 0, stream>>>(s_dst, EUU, Etot, 1, esc, nm, nden);
    gat_pass3<<<(Etot * HIDD + TPB - 1) / TPB, TPB, 0, stream>>>(s_src, s_dst, EUU, Etot, 1, big, esc, nden, acc);
    finalize_node<<<(NU * HIDD + TPB - 1) / TPB, TPB, 0, stream>>>(acc, bl, nullptr, dst,
                                                                   NU * HIDD, layer == 0 ? 1 : 0);
  }

  // ---- Phase 5: global attention pooling; `big` hosts Q/K/V ----
  float* Qb = big;
  float* Kb = big + (size_t)NU * HIDD;
  float* Vb = big + 2 * (size_t)NU * HIDD;
  launch_gemm(user, Wq, bq, nullptr, Qb, NU, HIDD, HIDD, 0, stream);
  launch_gemm(h2,   Wk, bk, nullptr, Kb, NU, HIDD, HIDD, 0, stream);
  launch_gemm(h2,   Wv, bv, nullptr, Vb, NU, HIDD, HIDD, 0, stream);
  dotrow<<<(NU + TPB - 1) / TPB, TPB, 0, stream>>>(Qb, Kb, sw);
  launch_fill(smax, 1, -__builtin_inff(), stream);
  launch_fill(smax + 1, 1, 0.f, stream);
  reduce_max_k<<<80, 256, 0, stream>>>(sw, NU, smax);
  expsum_k<<<(NU + TPB - 1) / TPB, TPB, 0, stream>>>(sw, NU, smax, smax + 1);
  gpool_k<<<HIDD, 256, 0, stream>>>(sw, Vb, smax + 1, gvec);
  matvec128<<<1, 128, 0, stream>>>(gvec, Wn, bn, gn);
  launch_gemm(user, Wm, bm, gn, userh, NU, HIDD, HIDD, 0, stream);  // + g broadcast

  // ---- Phase 6: output MLP over [user_h | g | dsl_embed] ----
  {
    dim3 grid((64 / 16 + 3) / 4, (NU + 31) / 32);
    gemm_cat_wmma<<<grid, 128, 0, stream>>>(userh, gn, dslE, Wo1, bo1, out1, NU, 64);
  }
  launch_gemm(out1, Wo2, bo2, nullptr, out, NU, 64, 64, 0, stream);
}
